// MultiSelfAttention_55551107007155
// MI455X (gfx1250) — matrix operand order repaired and checked by reading, not compiled, not run on hardware
//
#include <hip/hip_runtime.h>

// Problem constants (b,p,d)=(2,2048,768), h=12, n=64
#define B_  2
#define H_  12
#define P_  2048
#define D_  768
#define NH_ 64

typedef __attribute__((ext_vector_type(16))) _Float16 v16h;
typedef __attribute__((ext_vector_type(8)))  _Float16 v8h;
typedef __attribute__((ext_vector_type(8)))  float    v8f;

// D = A(16x32 f16) * B(32x16 f16) + C(16x16 f32)
__device__ __forceinline__ v8f wmma16(v16h a, v16h b, v8f c) {
  return __builtin_amdgcn_wmma_f32_16x16x32_f16(false, a, false, b, (short)0, c,
                                                false, false);
}

// A-fragment (16x32 f16): lane holds row m=lane&15, kg=lane>>4.
// halves j=0..7 -> K = 8*kg + j ; j=8..15 -> K = 16 + 8*kg + (j-8).
__device__ __forceinline__ v16h load_a_frag(const _Float16* rowbase, int kg) {
  v8h lo = *(const v8h*)(rowbase + 8 * kg);
  v8h hi = *(const v8h*)(rowbase + 16 + 8 * kg);
  v16h a;
#pragma unroll
  for (int i = 0; i < 8; ++i) { a[i] = lo[i]; a[i + 8] = hi[i]; }
  return a;
}

// B-fragment (32x16 f16): lane holds col n=lane&15, kg=lane>>4.
// halves j=0..7 -> K = 8*kg + j ; j=8..15 -> K = 16 + 8*kg + (j-8).
__device__ __forceinline__ v16h load_b_frag(const _Float16* rowbase, int kg) {
  v8h lo = *(const v8h*)(rowbase + 8 * kg);
  v8h hi = *(const v8h*)(rowbase + 16 + 8 * kg);
  v16h b;
#pragma unroll
  for (int i = 0; i < 8; ++i) { b[i] = lo[i]; b[i + 8] = hi[i]; }
  return b;
}

// CDNA5 async global->LDS copy (ASYNCcnt-tracked). One instruction moves
// 16B/lane. LDS dest: low 32 bits of the generic pointer are the LDS offset.
__device__ __forceinline__ void async_copy_b128(const _Float16* g,
                                                _Float16* l) {
  asm volatile("global_load_async_to_lds_b128 %0, %1, off"
               :
               : "v"((unsigned)(unsigned long long)l), "v"(g)
               : "memory");
}
#define ASYNC_WAIT_0() asm volatile("s_wait_asynccnt 0x0" ::: "memory")
#define ASYNC_WAIT_2() asm volatile("s_wait_asynccnt 0x2" ::: "memory")

// ---------------- cast / transpose kernels ----------------

__global__ void cvt_x_kernel(const float* __restrict__ in,
                             _Float16* __restrict__ out, int n) {
  int i = blockIdx.x * 256 + threadIdx.x;
  if (i < n) out[i] = (_Float16)in[i];
}

// w_qkv (h,3,d,n) f32 -> (h,3,n,d) f16
__global__ void cvt_wqkv_kernel(const float* __restrict__ in,
                                _Float16* __restrict__ out) {
  int i = blockIdx.x * 256 + threadIdx.x;
  if (i >= H_ * 3 * D_ * NH_) return;
  int n = i % NH_;
  int t = i / NH_;
  int d = t % D_;
  int ha = t / D_;
  out[((size_t)(ha * NH_ + n)) * D_ + d] = (_Float16)in[i];
}

// w_o (i,j) f32 -> (j,i) f16
__global__ void cvt_wo_kernel(const float* __restrict__ in,
                              _Float16* __restrict__ out) {
  int i = blockIdx.x * 256 + threadIdx.x;
  if (i >= D_ * D_) return;
  int c = i % D_;
  int r = i / D_;
  out[(size_t)c * D_ + r] = (_Float16)in[i];
}

// ---------------- QKV projection ----------------
// Y(2048x64) = X(2048x768) @ W(768x64) per (b,h,a); register-pipelined.
__global__ void __launch_bounds__(256)
qkv_gemm_kernel(const _Float16* __restrict__ xh,
                const _Float16* __restrict__ wqkvt,
                _Float16* __restrict__ Qh, _Float16* __restrict__ Kh,
                _Float16* __restrict__ Vt) {
  const int lane = threadIdx.x & 31;
  const int wave = threadIdx.x >> 5;
  const int ln = lane & 15, kg = lane >> 4, half = kg;

  int blk = blockIdx.x;  // b*H*3*(P/128)
  const int PT = P_ / 128;
  const int pt = blk % PT; blk /= PT;
  const int a  = blk % 3;  blk /= 3;
  const int h  = blk % H_; blk /= H_;
  const int b  = blk;
  const int p0 = pt * 128 + wave * 16;

  const _Float16* xrow = xh + ((size_t)(b * P_ + p0 + ln)) * D_;
  const _Float16* w0 = wqkvt + ((size_t)((h * 3 + a) * NH_) + 0 * 16 + ln) * D_;
  const _Float16* w1 = w0 + (size_t)16 * D_;
  const _Float16* w2 = w0 + (size_t)32 * D_;
  const _Float16* w3 = w0 + (size_t)48 * D_;

  v8f c0 = {}, c1 = {}, c2 = {}, c3 = {};
  v16h av = load_a_frag(xrow, kg);
  v16h b0 = load_b_frag(w0, kg);
  v16h b1 = load_b_frag(w1, kg);
  v16h b2 = load_b_frag(w2, kg);
  v16h b3 = load_b_frag(w3, kg);
  for (int kk = 32; kk < D_; kk += 32) {
    v16h avn = load_a_frag(xrow + kk, kg);
    v16h b0n = load_b_frag(w0 + kk, kg);
    v16h b1n = load_b_frag(w1 + kk, kg);
    v16h b2n = load_b_frag(w2 + kk, kg);
    v16h b3n = load_b_frag(w3 + kk, kg);
    c0 = wmma16(av, b0, c0);
    c1 = wmma16(av, b1, c1);
    c2 = wmma16(av, b2, c2);
    c3 = wmma16(av, b3, c3);
    av = avn; b0 = b0n; b1 = b1n; b2 = b2n; b3 = b3n;
  }
  c0 = wmma16(av, b0, c0);
  c1 = wmma16(av, b1, c1);
  c2 = wmma16(av, b2, c2);
  c3 = wmma16(av, b3, c3);

  v8f cs[4] = {c0, c1, c2, c3};
  const int bh = b * H_ + h;
  if (a < 2) {
    const float scale = (a == 0) ? 0.125f : 1.0f;  // fold 1/sqrt(64) into Q
    _Float16* dst = (a == 0 ? Qh : Kh) + ((size_t)bh * P_ + p0) * NH_;
#pragma unroll
    for (int f = 0; f < 4; ++f)
#pragma unroll
      for (int v = 0; v < 8; ++v)
        dst[(size_t)(v + 8 * half) * NH_ + f * 16 + ln] =
            (_Float16)(cs[f][v] * scale);
  } else {
    _Float16* dst = Vt + (size_t)bh * NH_ * P_;  // V stored transposed [64][P]
#pragma unroll
    for (int f = 0; f < 4; ++f) {
      v8h o;
#pragma unroll
      for (int v = 0; v < 8; ++v) o[v] = (_Float16)cs[f][v];
      *(v8h*)(dst + (size_t)(f * 16 + ln) * P_ + p0 + 8 * half) = o;
    }
  }
}

// ---------------- flash attention ----------------
// One wave = 16 query rows; key tiles of 32. K/V tiles staged block-wide into
// LDS via async-to-LDS copies, double buffered; online softmax; O = P@V.
#define KPAD 72  // 32x64 K tile rows padded 64->72 halves (bank stride 36)
#define VPAD 40  // 64x32 V tile rows padded 32->40 halves (bank stride 20)

__global__ void __launch_bounds__(256)
attn_kernel(const _Float16* __restrict__ Qh, const _Float16* __restrict__ Kh,
            const _Float16* __restrict__ Vt, _Float16* __restrict__ ao) {
  __shared__ _Float16 Kbuf[2][32][KPAD];
  __shared__ _Float16 Vbuf[2][64][VPAD];
  __shared__ _Float16 Pl[8][16][VPAD];

  const int tid = threadIdx.x;
  const int lane = tid & 31;
  const int wave = tid >> 5;
  const int ln = lane & 15, kg = lane >> 4, half = kg;

  int blk = blockIdx.x;  // b*H*(P/128)
  const int QT = P_ / 128;
  const int qt = blk % QT; blk /= QT;
  const int h  = blk % H_;
  const int b  = blk / H_;
  const int bh = b * H_ + h;
  const int p0 = qt * 128 + wave * 16;

  const _Float16* qrow = Qh + ((size_t)bh * P_ + p0 + ln) * NH_;
  const _Float16* kb   = Kh + (size_t)bh * P_ * NH_;
  const _Float16* vb   = Vt + (size_t)bh * NH_ * P_;

  // per-thread tile-copy assignment (block copies K:4KB + V:4KB per tile)
  const int kkey = tid >> 3, kc = (tid & 7) * 8;   // K: 32 rows x 64 halves
  const int vft = tid >> 2, vc = (tid & 3) * 8;    // V: 64 rows x 32 halves

  v16h qa0 = load_a_frag(qrow + 0, kg);
  v16h qa1 = load_a_frag(qrow + 32, kg);

  v8f o0 = {}, o1 = {}, o2 = {}, o3 = {};
  float mi[8], li[8];
#pragma unroll
  for (int v = 0; v < 8; ++v) { mi[v] = -1e30f; li[v] = 0.0f; }

  const float L2E = 1.44269504f;
  const int NT = P_ / 32;

  // prologue: stage tile 0 into buffer 0
  async_copy_b128(kb + (size_t)kkey * NH_ + kc, &Kbuf[0][kkey][kc]);
  async_copy_b128(vb + (size_t)vft * P_ + vc, &Vbuf[0][vft][vc]);

  for (int it = 0; it < NT; ++it) {
    const int cur = it & 1;
    if (it + 1 < NT) {  // prefetch next tile into the other buffer
      const int kt1 = (it + 1) * 32;
      async_copy_b128(kb + (size_t)(kt1 + kkey) * NH_ + kc,
                      &Kbuf[cur ^ 1][kkey][kc]);
      async_copy_b128(vb + (size_t)vft * P_ + kt1 + vc,
                      &Vbuf[cur ^ 1][vft][vc]);
      ASYNC_WAIT_2();  // this wave's tile-`it` copies (in-order) are done
    } else {
      ASYNC_WAIT_0();
    }
    __syncthreads();  // all waves' tile-`it` copies landed in LDS

    // S(16x32) = Q(16x64) @ K^T from LDS
    v16h kb00 = load_b_frag(&Kbuf[cur][ln][0], kg);
    v16h kb01 = load_b_frag(&Kbuf[cur][ln][32], kg);
    v16h kb10 = load_b_frag(&Kbuf[cur][16 + ln][0], kg);
    v16h kb11 = load_b_frag(&Kbuf[cur][16 + ln][32], kg);
    v8f s0 = {}, s1 = {};
    s0 = wmma16(qa0, kb00, s0);
    s0 = wmma16(qa1, kb01, s0);
    s1 = wmma16(qa0, kb10, s1);
    s1 = wmma16(qa1, kb11, s1);

    // online softmax; row m = v + 8*half lives in one 16-lane half per v
    float al[8];
#pragma unroll
    for (int v = 0; v < 8; ++v) {
      float rm = fmaxf(s0[v], s1[v]);
      rm = fmaxf(rm, __shfl_xor(rm, 1));
      rm = fmaxf(rm, __shfl_xor(rm, 2));
      rm = fmaxf(rm, __shfl_xor(rm, 4));
      rm = fmaxf(rm, __shfl_xor(rm, 8));
      float nm = fmaxf(mi[v], rm);
      al[v] = exp2f((mi[v] - nm) * L2E);
      float e0 = exp2f((s0[v] - nm) * L2E);
      float e1 = exp2f((s1[v] - nm) * L2E);
      s0[v] = e0; s1[v] = e1;
      float rs = e0 + e1;
      rs += __shfl_xor(rs, 1);
      rs += __shfl_xor(rs, 2);
      rs += __shfl_xor(rs, 4);
      rs += __shfl_xor(rs, 8);
      li[v] = al[v] * li[v] + rs;
      mi[v] = nm;
    }
#pragma unroll
    for (int v = 0; v < 8; ++v) {
      o0[v] *= al[v]; o1[v] *= al[v]; o2[v] *= al[v]; o3[v] *= al[v];
    }

    // P: C-fragment layout -> per-wave LDS tile -> A-fragment layout.
    // LDS ops from one wave complete in order (ISA 7.3), so only a compiler
    // fence is needed between the stores and the re-load.
#pragma unroll
    for (int v = 0; v < 8; ++v) {
      Pl[wave][v + 8 * half][ln]      = (_Float16)s0[v];
      Pl[wave][v + 8 * half][16 + ln] = (_Float16)s1[v];
    }
    asm volatile("" ::: "memory");
    __builtin_amdgcn_wave_barrier();
    v16h pa = load_a_frag(&Pl[wave][ln][0], kg);

    // O += P(16x32) @ V(32x64) from LDS (V staged transposed [feat][key])
    v16h vb0 = load_b_frag(&Vbuf[cur][0 * 16 + ln][0], kg);
    v16h vb1 = load_b_frag(&Vbuf[cur][1 * 16 + ln][0], kg);
    v16h vb2 = load_b_frag(&Vbuf[cur][2 * 16 + ln][0], kg);
    v16h vb3 = load_b_frag(&Vbuf[cur][3 * 16 + ln][0], kg);
    o0 = wmma16(pa, vb0, o0);
    o1 = wmma16(pa, vb1, o1);
    o2 = wmma16(pa, vb2, o2);
    o3 = wmma16(pa, vb3, o3);
    __syncthreads();  // everyone done with buffer `cur` before it is refilled
  }

  // normalize and store to ao[b][p][h*64+n] (f16)
  float inv[8];
#pragma unroll
  for (int v = 0; v < 8; ++v) inv[v] = 1.0f / li[v];
  v8f os[4] = {o0, o1, o2, o3};
  _Float16* abase = ao + ((size_t)(b * P_ + p0)) * D_ + h * NH_;
#pragma unroll
  for (int f = 0; f < 4; ++f)
#pragma unroll
    for (int v = 0; v < 8; ++v)
      abase[(size_t)(v + 8 * half) * D_ + f * 16 + ln] =
          (_Float16)(os[f][v] * inv[v]);
}

// ---------------- output projection ----------------
// out(4096x768 f32) = ao(4096x768 f16) @ w_o; wot is w_o transposed.
__global__ void __launch_bounds__(256)
proj_kernel(const _Float16* __restrict__ ao, const _Float16* __restrict__ wot,
            float* __restrict__ out) {
  const int lane = threadIdx.x & 31;
  const int wave = threadIdx.x >> 5;
  const int ln = lane & 15, kg = lane >> 4, half = kg;

  int blk = blockIdx.x;  // (4096/128) * (768/64)
  const int NT = D_ / 64;
  const int nt = blk % NT;
  const int mt = blk / NT;
  const int r0 = mt * 128 + wave * 16;
  const int c0 = nt * 64;

  const _Float16* arow = ao + (size_t)(r0 + ln) * D_;
  const _Float16* w0 = wot + (size_t)(c0 + 0 * 16 + ln) * D_;
  const _Float16* w1 = w0 + (size_t)16 * D_;
  const _Float16* w2 = w0 + (size_t)32 * D_;
  const _Float16* w3 = w0 + (size_t)48 * D_;

  v8f c0v = {}, c1v = {}, c2v = {}, c3v = {};
  v16h av = load_a_frag(arow, kg);
  v16h b0 = load_b_frag(w0, kg);
  v16h b1 = load_b_frag(w1, kg);
  v16h b2 = load_b_frag(w2, kg);
  v16h b3 = load_b_frag(w3, kg);
  for (int kk = 32; kk < D_; kk += 32) {
    v16h avn = load_a_frag(arow + kk, kg);
    v16h b0n = load_b_frag(w0 + kk, kg);
    v16h b1n = load_b_frag(w1 + kk, kg);
    v16h b2n = load_b_frag(w2 + kk, kg);
    v16h b3n = load_b_frag(w3 + kk, kg);
    c0v = wmma16(av, b0, c0v);
    c1v = wmma16(av, b1, c1v);
    c2v = wmma16(av, b2, c2v);
    c3v = wmma16(av, b3, c3v);
    av = avn; b0 = b0n; b1 = b1n; b2 = b2n; b3 = b3n;
  }
  c0v = wmma16(av, b0, c0v);
  c1v = wmma16(av, b1, c1v);
  c2v = wmma16(av, b2, c2v);
  c3v = wmma16(av, b3, c3v);

  v8f cs[4] = {c0v, c1v, c2v, c3v};
#pragma unroll
  for (int f = 0; f < 4; ++f)
#pragma unroll
    for (int v = 0; v < 8; ++v)
      out[(size_t)(r0 + v + 8 * half) * D_ + c0 + f * 16 + ln] = cs[f][v];
}

// ---------------- launcher ----------------

extern "C" void kernel_launch(void* const* d_in, const int* in_sizes, int n_in,
                              void* d_out, int out_size, void* d_ws,
                              size_t ws_size, hipStream_t stream) {
  const float* x     = (const float*)d_in[0];  // (2,2048,768)
  const float* w_qkv = (const float*)d_in[1];  // (12,3,768,64)
  const float* w_o   = (const float*)d_in[2];  // (768,768)
  float* out = (float*)d_out;                  // (2,2048,768) f32

  _Float16* ws = (_Float16*)d_ws;
  const size_t NX   = (size_t)B_ * P_ * D_;
  const size_t NW   = (size_t)H_ * 3 * D_ * NH_;
  const size_t NWO  = (size_t)D_ * D_;
  const size_t NQKV = (size_t)B_ * H_ * P_ * NH_;
  _Float16* xh    = ws;
  _Float16* wqkvt = xh + NX;
  _Float16* wot   = wqkvt + NW;
  _Float16* Qh    = wot + NWO;
  _Float16* Kh    = Qh + NQKV;
  _Float16* Vt    = Kh + NQKV;
  _Float16* ao    = Vt + NQKV;

  cvt_x_kernel<<<(int)(NX / 256), 256, 0, stream>>>(x, xh, (int)NX);
  cvt_wqkv_kernel<<<(int)(NW / 256), 256, 0, stream>>>(w_qkv, wqkvt);
  cvt_wo_kernel<<<(int)(NWO / 256), 256, 0, stream>>>(w_o, wot);

  qkv_gemm_kernel<<<B_ * H_ * 3 * (P_ / 128), 256, 0, stream>>>(xh, wqkvt, Qh,
                                                                Kh, Vt);
  attn_kernel<<<B_ * H_ * (P_ / 128), 256, 0, stream>>>(Qh, Kh, Vt, ao);
  proj_kernel<<<(B_ * P_ / 128) * (D_ / 64), 256, 0, stream>>>(ao, wot, out);
}
